// DeformableConvV2_73160472920182
// MI455X (gfx1250) — matrix operand order repaired and checked by reading, not compiled, not run on hardware
//
#include <hip/hip_runtime.h>

typedef float v4f  __attribute__((ext_vector_type(4)));
typedef float v8f  __attribute__((ext_vector_type(8)));
typedef __bf16 bf16;
typedef bf16 v8bf  __attribute__((ext_vector_type(8)));
typedef bf16 v16bf __attribute__((ext_vector_type(16)));
typedef unsigned int u32x4 __attribute__((ext_vector_type(4)));
typedef int i32x4 __attribute__((ext_vector_type(4)));
typedef int i32x8 __attribute__((ext_vector_type(8)));

#define B_    4
#define C_    256
#define H_    64
#define W_    64
#define O_    256
#define KK_   9
#define HW_   4096          // H_*W_
#define KDIM  2304          // KK_*C_
#define NPIX  16384         // B_*HW_

#if __has_builtin(__builtin_amdgcn_tensor_load_to_lds) && \
    __has_builtin(__builtin_amdgcn_s_wait_tensorcnt)
#define HAVE_TDM 1
#else
#define HAVE_TDM 0
#endif

// ---------------------------------------------------------------------------
// Kernel 1: offset conv  om[b][ch<27][ho][wo]  (3x3 conv over C=256 + bias)
// ---------------------------------------------------------------------------
__global__ __launch_bounds__(64) void offs_conv_kernel(
    const float* __restrict__ x, const float* __restrict__ w_off,
    const float* __restrict__ b_off, float* __restrict__ om) {
  const int wo = threadIdx.x;
  const int ho = blockIdx.x;
  const int b  = blockIdx.y;

  float acc[27];
#pragma unroll
  for (int i = 0; i < 27; ++i) acc[i] = b_off[i];

  const float* xb = x + b * (C_ * HW_);
  for (int c = 0; c < C_; ++c) {
    const float* xc = xb + c * HW_;
#pragma unroll
    for (int ky = 0; ky < 3; ++ky) {
      const int y = ho + ky - 1;
      if (y < 0 || y >= H_) continue;           // uniform per block
#pragma unroll
      for (int kx = 0; kx < 3; ++kx) {
        const int xx = wo + kx - 1;
        const float xv = (xx >= 0 && xx < W_) ? xc[y * W_ + xx] : 0.0f;
        const float* wp = w_off + (c * 9 + ky * 3 + kx);
#pragma unroll
        for (int oc = 0; oc < 27; ++oc)
          acc[oc] = fmaf(xv, wp[oc * (C_ * 9)], acc[oc]);
      }
    }
  }
  const int sp = ho * W_ + wo;
#pragma unroll
  for (int oc = 0; oc < 27; ++oc)
    om[(b * 27 + oc) * HW_ + sp] = acc[oc];
}

// ---------------------------------------------------------------------------
// Kernel 2: weight transpose + split-bf16
// ---------------------------------------------------------------------------
__global__ __launch_bounds__(256) void weight_prep_kernel(
    const float* __restrict__ w_dcn, bf16* __restrict__ Wh,
    bf16* __restrict__ Wl) {
  const int idx = blockIdx.x * 256 + threadIdx.x;   // 0 .. O_*KDIM-1
  const int o  = idx / KDIM;
  const int r  = idx % KDIM;
  const int kk = r / C_;
  const int c  = r % C_;
  const float v = w_dcn[(o * C_ + c) * KK_ + kk];
  const bf16 h = (bf16)v;
  const bf16 l = (bf16)(v - (float)h);
  Wh[idx] = h;
  Wl[idx] = l;
}

// ---------------------------------------------------------------------------
// Kernel 3: bilinear sampling + sigmoid mask -> split-bf16 im2col (transposed)
// ---------------------------------------------------------------------------
__global__ __launch_bounds__(64) void sample_kernel(
    const float* __restrict__ x, const float* __restrict__ om,
    bf16* __restrict__ Bh, bf16* __restrict__ Bl) {
  const int wo = threadIdx.x;
  const int ho = blockIdx.x;
  const int k  = blockIdx.y;
  const int b  = blockIdx.z;
  const int sp = ho * W_ + wo;

  const float* omb = om + b * 27 * HW_;
  const float off_y = omb[(2 * k)     * HW_ + sp];
  const float off_x = omb[(2 * k + 1) * HW_ + sp];
  const float mraw  = omb[(18 + k)    * HW_ + sp];
  const float mask  = 1.0f / (1.0f + __expf(-mraw));

  const float ys = (float)(ho - 1 + k / 3) + off_y;
  const float xs = (float)(wo - 1 + k % 3) + off_x;
  const float y0f = floorf(ys), x0f = floorf(xs);
  const float wy1 = ys - y0f, wy0 = 1.0f - wy1;
  const float wx1 = xs - x0f, wx0 = 1.0f - wx1;
  const int y0 = (int)y0f, x0 = (int)x0f;
  const int y1 = y0 + 1,   x1 = x0 + 1;

  const float vy0 = (y0 >= 0 && y0 < H_) ? 1.0f : 0.0f;
  const float vy1 = (y1 >= 0 && y1 < H_) ? 1.0f : 0.0f;
  const float vx0 = (x0 >= 0 && x0 < W_) ? 1.0f : 0.0f;
  const float vx1 = (x1 >= 0 && x1 < W_) ? 1.0f : 0.0f;

  const float w00 = mask * wy0 * wx0 * vy0 * vx0;
  const float w01 = mask * wy0 * wx1 * vy0 * vx1;
  const float w10 = mask * wy1 * wx0 * vy1 * vx0;
  const float w11 = mask * wy1 * wx1 * vy1 * vx1;

  const int yc0 = min(max(y0, 0), H_ - 1), yc1 = min(max(y1, 0), H_ - 1);
  const int xc0 = min(max(x0, 0), W_ - 1), xc1 = min(max(x1, 0), W_ - 1);
  const int i00 = yc0 * W_ + xc0, i01 = yc0 * W_ + xc1;
  const int i10 = yc1 * W_ + xc0, i11 = yc1 * W_ + xc1;

  const float* xb = x + b * (C_ * HW_);
  const size_t n = (size_t)b * HW_ + sp;            // pixel row
  bf16* dh = Bh + n * KDIM + k * C_;
  bf16* dl = Bl + n * KDIM + k * C_;

  alignas(16) bf16 hbuf[8];
  alignas(16) bf16 lbuf[8];
  for (int c = 0; c < C_; ++c) {
    const float* xc = xb + c * HW_;
    const float g = w00 * xc[i00] + w01 * xc[i01] +
                    w10 * xc[i10] + w11 * xc[i11];
    const bf16 h = (bf16)g;
    const bf16 l = (bf16)(g - (float)h);
    hbuf[c & 7] = h;
    lbuf[c & 7] = l;
    if ((c & 7) == 7) {                             // 16B packed stores
      *(v8bf*)(dh + (c & ~7)) = *(const v8bf*)hbuf;
      *(v8bf*)(dl + (c & ~7)) = *(const v8bf*)lbuf;
    }
  }
}

// ---------------------------------------------------------------------------
// Kernel 4: split-bf16 GEMM with TDM double-buffered LDS staging.
//   out[o][n] = sum_k W[o][k] * B[n][k];  acc += Ah*Bh + Ah*Bl + Al*Bh
// Block 256 thr (8 waves), tile 64(M)x128(N); wave 32x32 (2x2 WMMA tiles).
// ---------------------------------------------------------------------------
#define MBLK 64
#define NBLK 128
#define KCH2 32
#define LPAD 40   // LDS row stride in bf16: 64B row + 16B pad (TDM pad 4dw/16dw)

__device__ inline v16bf ld_frag(const bf16* rowbase, int off0, int off1) {
  const v8bf p0 = *(const v8bf*)(rowbase + off0);
  const v8bf p1 = *(const v8bf*)(rowbase + off1);
  return __builtin_shufflevector(p0, p1, 0, 1, 2, 3, 4, 5, 6, 7,
                                 8, 9, 10, 11, 12, 13, 14, 15);
}

#if HAVE_TDM
// Issue one 2D TDM tile load: rows x cols(bf16) with 64B->80B LDS row padding.
// D# layout per CDNA5 ISA ch.8: group0 {count|lds_addr|global_addr|type=2},
// group1 {data_size=2B, pad_enable, pad_interval=3(16dw), pad_amount=3(4dw),
//         tensor_dim0/1, tile_dim0/1, tensor_dim0_stride}.
__device__ inline void tdm_load_2d(unsigned int lds_addr, const bf16* gptr,
                                   unsigned int tdim0, unsigned int tdim1,
                                   unsigned int tile0, unsigned int tile1,
                                   unsigned int stride0) {
  const unsigned long long ga = (unsigned long long)(uintptr_t)gptr;
  u32x4 g0;
  g0[0] = 1u;                                       // count=1, user descriptor
  g0[1] = lds_addr;                                 // LDS byte address
  g0[2] = (unsigned int)ga;                         // global_addr[31:0]
  g0[3] = (unsigned int)(ga >> 32) | (2u << 30);    // addr[56:32] | type=2
  i32x8 g1;
  g1[0] = (1 << 16) | (1 << 20) | (3 << 22) | (3 << 25);
  g1[1] = (int)((tdim0 & 0xFFFFu) << 16);
  g1[2] = (int)((tdim0 >> 16) | ((tdim1 & 0xFFFFu) << 16));
  g1[3] = (int)((tdim1 >> 16) | (tile0 << 16));
  g1[4] = (int)(tile1 & 0xFFFFu);
  g1[5] = (int)stride0;
  g1[6] = 0;
  g1[7] = 0;
  const i32x4 gz4 = {0, 0, 0, 0};
  const i32x8 gz8 = {0, 0, 0, 0, 0, 0, 0, 0};
  __builtin_amdgcn_tensor_load_to_lds(g0, g1, gz4, gz4, gz8, 0);
}
#endif

__global__ __launch_bounds__(256) void dcn_gemm_kernel(
    const bf16* __restrict__ Ah_g,  // [O_][KDIM] hi
    const bf16* __restrict__ Al_g,  // [O_][KDIM] lo
    const bf16* __restrict__ Bh_g,  // [NPIX][KDIM] hi
    const bf16* __restrict__ Bl_g,  // [NPIX][KDIM] lo
    float* __restrict__ out) {      // [B_][O_][HW_]
  __shared__ bf16 Ah[2][MBLK][LPAD], Al[2][MBLK][LPAD];
  __shared__ bf16 Bh[2][NBLK][LPAD], Bl[2][NBLK][LPAD];

  const int tid  = threadIdx.x;
  const int lane = tid & 31;
  const int wave = tid >> 5;
  const int wm   = wave >> 2;          // 0..1  (M dir)
  const int wn   = wave & 3;           // 0..3  (N dir)
  const int half = lane >> 4;          // lane half-select (ISA 16-bit layouts)
  const int l15  = lane & 15;
  const int m0   = blockIdx.x * MBLK;
  const int n0   = blockIdx.y * NBLK;

  const int koffA = half * 8;          // A: lanes<16 K0-7/16-23, >=16 K8-15/24-31
  const int koffB = half * 8;          // B: lanes<16 K0-7/16-23, >=16 K8-15/24-31

  const v8f zero = {0.f, 0.f, 0.f, 0.f, 0.f, 0.f, 0.f, 0.f};
  v8f acc[2][2];
#pragma unroll
  for (int i = 0; i < 2; ++i)
#pragma unroll
    for (int j = 0; j < 2; ++j) acc[i][j] = zero;

  auto compute_stage = [&](int buf) {
    v16bf afh[2], afl[2], bfh[2], bfl[2];
#pragma unroll
    for (int s = 0; s < 2; ++s) {
      const int arow = wm * 32 + s * 16 + l15;
      afh[s] = ld_frag(&Ah[buf][arow][0], koffA, 16 + koffA);
      afl[s] = ld_frag(&Al[buf][arow][0], koffA, 16 + koffA);
      const int bcol = wn * 32 + s * 16 + l15;
      bfh[s] = ld_frag(&Bh[buf][bcol][0], koffB, 16 + koffB);
      bfl[s] = ld_frag(&Bl[buf][bcol][0], koffB, 16 + koffB);
    }
#pragma unroll
    for (int i = 0; i < 2; ++i)
#pragma unroll
      for (int j = 0; j < 2; ++j) {
        acc[i][j] = __builtin_amdgcn_wmma_f32_16x16x32_bf16(
            false, afl[i], false, bfh[j], (short)0, acc[i][j], false, false);
        acc[i][j] = __builtin_amdgcn_wmma_f32_16x16x32_bf16(
            false, afh[i], false, bfl[j], (short)0, acc[i][j], false, false);
        acc[i][j] = __builtin_amdgcn_wmma_f32_16x16x32_bf16(
            false, afh[i], false, bfh[j], (short)0, acc[i][j], false, false);
      }
  };

#if HAVE_TDM
  // ---- TDM double-buffered pipeline: wave 0 drives the Tensor Data Mover ---
  const bool issuer = (wave == 0);
  const unsigned int lAh[2] = {(unsigned int)(uintptr_t)&Ah[0][0][0],
                               (unsigned int)(uintptr_t)&Ah[1][0][0]};
  const unsigned int lAl[2] = {(unsigned int)(uintptr_t)&Al[0][0][0],
                               (unsigned int)(uintptr_t)&Al[1][0][0]};
  const unsigned int lBh[2] = {(unsigned int)(uintptr_t)&Bh[0][0][0],
                               (unsigned int)(uintptr_t)&Bh[1][0][0]};
  const unsigned int lBl[2] = {(unsigned int)(uintptr_t)&Bl[0][0][0],
                               (unsigned int)(uintptr_t)&Bl[1][0][0]};

  auto issue_stage = [&](int buf, int k0) {
    tdm_load_2d(lAh[buf], Ah_g + (size_t)m0 * KDIM + k0, KDIM, O_,   KCH2, MBLK, KDIM);
    tdm_load_2d(lAl[buf], Al_g + (size_t)m0 * KDIM + k0, KDIM, O_,   KCH2, MBLK, KDIM);
    tdm_load_2d(lBh[buf], Bh_g + (size_t)n0 * KDIM + k0, KDIM, NPIX, KCH2, NBLK, KDIM);
    tdm_load_2d(lBl[buf], Bl_g + (size_t)n0 * KDIM + k0, KDIM, NPIX, KCH2, NBLK, KDIM);
  };

  const int NCH = KDIM / KCH2;                 // 72 chunks
  if (issuer) issue_stage(0, 0);
  for (int i = 0; i < NCH; ++i) {
    const int buf = i & 1;
    __syncthreads();                           // prev compute on buf^1 finished
    if (issuer) {
      if (i + 1 < NCH) {
        issue_stage(buf ^ 1, (i + 1) * KCH2);  // prefetch next stage
        __builtin_amdgcn_s_wait_tensorcnt(4);  // in-order: current 4 retired
      } else {
        __builtin_amdgcn_s_wait_tensorcnt(0);
      }
    }
    __syncthreads();                           // publish buf
    compute_stage(buf);
  }
#else
  // ---- fallback: cooperative vector loads, single buffer ----
  for (int k0 = 0; k0 < KDIM; k0 += KCH2) {
    {
      const int r   = tid >> 2;
      const int seg = (tid & 3) * 8;
      const size_t g = (size_t)(m0 + r) * KDIM + k0 + seg;
      *(v8bf*)&Ah[0][r][seg] = *(const v8bf*)(Ah_g + g);
      *(v8bf*)&Al[0][r][seg] = *(const v8bf*)(Al_g + g);
    }
    {
      const int n   = tid >> 1;
      const int seg = (tid & 1) * 16;
      const size_t g = (size_t)(n0 + n) * KDIM + k0 + seg;
      *(v8bf*)&Bh[0][n][seg]     = *(const v8bf*)(Bh_g + g);
      *(v8bf*)&Bh[0][n][seg + 8] = *(const v8bf*)(Bh_g + g + 8);
      *(v8bf*)&Bl[0][n][seg]     = *(const v8bf*)(Bl_g + g);
      *(v8bf*)&Bl[0][n][seg + 8] = *(const v8bf*)(Bl_g + g + 8);
    }
    __syncthreads();
    compute_stage(0);
    __syncthreads();
  }
#endif

  // --- epilogue: D VGPR r holds rows (r, r+8); lanes 0-15 / 16-31 split ---
#pragma unroll
  for (int i = 0; i < 2; ++i) {
#pragma unroll
    for (int j = 0; j < 2; ++j) {
      const int mb = m0 + wm * 32 + i * 16 + half * 8;
      const int n  = n0 + wn * 32 + j * 16 + l15;
      const int b  = n >> 12;              // pixel -> batch (HW_ = 4096)
      const int p  = n & (HW_ - 1);
      float* dst = out + (size_t)b * O_ * HW_ + p;
#pragma unroll
      for (int r = 0; r < 8; ++r) {
        const int o = mb + r;
        dst[(size_t)o * HW_] = acc[i][j][r];
      }
    }
  }
}

// ---------------------------------------------------------------------------
extern "C" void kernel_launch(void* const* d_in, const int* in_sizes, int n_in,
                              void* d_out, int out_size, void* d_ws, size_t ws_size,
                              hipStream_t stream) {
  const float* x     = (const float*)d_in[0];
  const float* w_off = (const float*)d_in[1];
  const float* b_off = (const float*)d_in[2];
  const float* w_dcn = (const float*)d_in[3];
  float* out = (float*)d_out;

  char* ws = (char*)d_ws;
  float* om = (float*)(ws);                          //  1,769,472 B
  bf16* Wh  = (bf16*)(ws + 1769472);                 //  1,179,648 B
  bf16* Wl  = (bf16*)(ws + 1769472 + 1179648);       //  1,179,648 B
  bf16* Bh  = (bf16*)(ws + 4128768);                 // 75,497,472 B
  bf16* Bl  = (bf16*)(ws + 4128768 + 75497472);      // 75,497,472 B

  offs_conv_kernel<<<dim3(H_, B_), 64, 0, stream>>>(x, w_off, b_off, om);
  weight_prep_kernel<<<(O_ * KDIM) / 256, 256, 0, stream>>>(w_dcn, Wh, Wl);
  sample_kernel<<<dim3(H_, KK_, B_), 64, 0, stream>>>(x, om, Bh, Bl);
  dcn_gemm_kernel<<<dim3(O_ / MBLK, NPIX / NBLK), 256, 0, stream>>>(
      Wh, Wl, Bh, Bl, out);
}
